// BertCrf_59201829208230
// MI455X (gfx1250) — matrix operand order repaired and checked by reading, not compiled, not run on hardware
//
#include <hip/hip_runtime.h>
#include <hip/hip_bf16.h>
#include <math.h>

typedef __attribute__((ext_vector_type(16))) _Float16 v16h;
typedef __attribute__((ext_vector_type(8)))  float    v8f;

#define HD   768
#define LN   9
#define SQ   512
#define BT   64
#define KBLK 24      // 768 / 32

// ---------------------------------------------------------------------------
// Emission GEMM: emis[B*S, 9] = hidden[B*S, 768] @ W[768, 9] + bias
// One wave per 16-row tile; W staged in LDS as pre-swizzled WMMA B-fragments.
// ---------------------------------------------------------------------------
__global__ __launch_bounds__(256) void emis_gemm(const float* __restrict__ hidden,
                                                 const float* __restrict__ W,
                                                 const float* __restrict__ bias,
                                                 float* __restrict__ emis)
{
    __shared__ __align__(32) _Float16 bfrag[KBLK * 32 * 16];

    // Cooperative fill of B fragments: element e of lane l in block blk maps to
    // K = blk*32 + (l>>4)*8 + (e<8 ? e : e+8), N = l&15 (zero-padded for N >= 9).
    for (int i = threadIdx.x; i < KBLK * 32 * 16; i += 256) {
        int blk  = i >> 9;
        int rem  = i & 511;
        int lane = rem >> 4;
        int e    = rem & 15;
        int n    = lane & 15;
        int k    = blk * 32 + ((lane >> 4) << 3) + (e < 8 ? e : e + 8);
        float v  = (n < LN) ? W[k * LN + n] : 0.0f;
        bfrag[i] = (_Float16)v;
    }
    __syncthreads();

    int wave = threadIdx.x >> 5;
    int lane = threadIdx.x & 31;
    int m0   = (blockIdx.x * 8 + wave) * 16;
    int row  = m0 + (lane & 15);
    int aoff = (lane >> 4) << 3;             // lanes 16-31 start at K+8
    const float* hrow = hidden + (size_t)row * HD + aoff;

    v8f c = {};
    for (int blk = 0; blk < KBLK; ++blk) {
        if (blk + 2 < KBLK)
            __builtin_prefetch(hrow + (blk + 2) * 32, 0, 3);   // WGP-scope prefetch
        const float4* p = (const float4*)(hrow + blk * 32);
        float4 x0 = p[0], x1 = p[1], x2 = p[4], x3 = p[5];     // K+0..7, K+16..23
        v16h a;
        a[0] = (_Float16)x0.x;  a[1] = (_Float16)x0.y;
        a[2] = (_Float16)x0.z;  a[3] = (_Float16)x0.w;
        a[4] = (_Float16)x1.x;  a[5] = (_Float16)x1.y;
        a[6] = (_Float16)x1.z;  a[7] = (_Float16)x1.w;
        a[8] = (_Float16)x2.x;  a[9] = (_Float16)x2.y;
        a[10]= (_Float16)x2.z;  a[11]= (_Float16)x2.w;
        a[12]= (_Float16)x3.x;  a[13]= (_Float16)x3.y;
        a[14]= (_Float16)x3.z;  a[15]= (_Float16)x3.w;
        v16h bm = *(const v16h*)&bfrag[(blk * 32 + lane) * 16];
        c = __builtin_amdgcn_wmma_f32_16x16x32_f16(false, a, false, bm,
                                                   (short)0, c, false, false);
    }

    // C layout: N = lane&15, M = r + (lane>>4)*8 for VGPR r.
    int n = lane & 15;
    if (n < LN) {
        float bn = bias[n];
        int   mb = m0 + ((lane >> 4) << 3);
        #pragma unroll
        for (int r = 0; r < 8; ++r)
            emis[(size_t)(mb + r) * LN + n] = c[r] + bn;
    }
}

// ------------------------- tree reductions over 9 -------------------------
__device__ __forceinline__ float max9(const float* v)
{
    float m01 = fmaxf(v[0], v[1]), m23 = fmaxf(v[2], v[3]);
    float m45 = fmaxf(v[4], v[5]), m67 = fmaxf(v[6], v[7]);
    return fmaxf(fmaxf(fmaxf(m01, m23), fmaxf(m45, m67)), v[8]);
}

__device__ __forceinline__ float lse9(const float* v)   // logsumexp, depth-4 trees
{
    float mx = max9(v);
    float e0 = __expf(v[0] - mx), e1 = __expf(v[1] - mx);
    float e2 = __expf(v[2] - mx), e3 = __expf(v[3] - mx);
    float e4 = __expf(v[4] - mx), e5 = __expf(v[5] - mx);
    float e6 = __expf(v[6] - mx), e7 = __expf(v[7] - mx);
    float e8 = __expf(v[8] - mx);
    float s = ((e0 + e1) + (e2 + e3)) + ((e4 + e5) + (e6 + e7)) + e8;
    return mx + __logf(s);
}

// first-index argmax over 9 via tree ('>=' keeps the lower index on ties)
__device__ __forceinline__ void argmax9(const float* v, float* bestv, int* besti)
{
    float v0 = v[0]; int i0 = 0;
    float v1 = v[2]; int i1 = 2;
    float v2 = v[4]; int i2 = 4;
    float v3 = v[6]; int i3 = 6;
    if (!(v0 >= v[1])) { v0 = v[1]; i0 = 1; }
    if (!(v1 >= v[3])) { v1 = v[3]; i1 = 3; }
    if (!(v2 >= v[5])) { v2 = v[5]; i2 = 5; }
    if (!(v3 >= v[7])) { v3 = v[7]; i3 = 7; }
    if (!(v0 >= v1)) { v0 = v1; i0 = i1; }
    if (!(v2 >= v3)) { v2 = v3; i2 = i3; }
    if (!(v0 >= v2)) { v0 = v2; i0 = i2; }
    if (!(v0 >= v[8])) { v0 = v[8]; i0 = 8; }
    *bestv = v0; *besti = i0;
}

// ---------------------------------------------------------------------------
// Merged CRF scan kernel: blocks [0,64) run the forward algorithm (logZ +
// gold-path score), blocks [64,128) run Viterbi + backtrack. Running both
// scan families concurrently halves the serial-scan wall time.
// One wave per batch; lane j owns state for label j (j < 9).
// ---------------------------------------------------------------------------
__global__ __launch_bounds__(32) void crf_scan(const float* __restrict__ emis,
                                               const int*   __restrict__ label,
                                               const int*   __restrict__ mask,
                                               const float* __restrict__ startT,
                                               const float* __restrict__ endT,
                                               const float* __restrict__ trans,
                                               float* __restrict__ out_predict,
                                               float* __restrict__ accum)
{
    __shared__ unsigned char bp[(SQ - 1) * LN];

    int b    = blockIdx.x & (BT - 1);
    int lane = threadIdx.x;
    int j    = (lane < LN) ? lane : 0;

    float tc[LN];                                  // transitions column j
    #pragma unroll
    for (int i = 0; i < LN; ++i) tc[i] = trans[i * LN + j];

    const float* em_b = emis + (size_t)b * SQ * LN;

    if (blockIdx.x < BT) {
        // ------------------ forward algorithm + numerator ------------------
        float alpha = startT[j] + em_b[j];         // t = 0

        float score = 0.0f;
        int   prev  = label[b * SQ];
        if (lane == 0) score = startT[prev] + em_b[prev];

        for (int t = 1; t < SQ; ++t) {
            float v[LN];
            #pragma unroll
            for (int i = 0; i < LN; ++i) v[i] = __shfl(alpha, i, 32) + tc[i];

            float mt = (float)mask[b * SQ + t];
            const float* em_t = em_b + t * LN;

            float nxt = lse9(v) + em_t[j];
            alpha = (mt > 0.0f) ? nxt : alpha;

            if (lane == 0) {
                int tg = label[b * SQ + t];
                score += (trans[prev * LN + tg] + em_t[tg]) * mt;
                prev = (mt > 0.0f) ? tg : prev;
            }
        }

        // logZ = logsumexp_j(alpha_j + end_j); gather with all lanes active.
        float fin = alpha + endT[j];
        float vals[LN];
        #pragma unroll
        for (int i = 0; i < LN; ++i) vals[i] = __shfl(fin, i, 32);
        if (lane == 0) {
            float logZ = lse9(vals);
            score += endT[prev];
            atomicAdd(accum, score - logZ);        // sum of per-batch llh
        }
    } else {
        // --------------------- Viterbi decode + backtrack ------------------
        float v = startT[j] + em_b[j];

        for (int t = 1; t < SQ; ++t) {
            float x[LN];
            #pragma unroll
            for (int i = 0; i < LN; ++i) x[i] = __shfl(v, i, 32) + tc[i];
            float best; int arg;
            argmax9(x, &best, &arg);
            v = best + em_b[t * LN + j];
            if (lane < LN) bp[(t - 1) * LN + j] = (unsigned char)arg;
        }

        float fin = v + endT[j];
        float vals[LN];
        #pragma unroll
        for (int i = 0; i < LN; ++i) vals[i] = __shfl(fin, i, 32);
        __syncthreads();                           // bp visible to lane 0

        if (lane == 0) {
            float best; int tag;
            argmax9(vals, &best, &tag);

            int corr = 0;
            {
                int lab  = label[b * SQ + (SQ - 1)];
                int pred = (lab > 0) ? tag : 0;
                out_predict[b * SQ + (SQ - 1)] = (float)pred;
                corr += (pred == lab);
            }
            for (int t = SQ - 2; t >= 0; --t) {
                tag = bp[t * LN + tag];
                int lab  = label[b * SQ + t];
                int pred = (lab > 0) ? tag : 0;
                out_predict[b * SQ + t] = (float)pred;
                corr += (pred == lab);
            }
            atomicAdd(accum + 1, (float)corr);
        }
    }
}

// ---------------------------------------------------------------------------
__global__ void init_accum(float* accum)
{
    if (threadIdx.x < 2) accum[threadIdx.x] = 0.0f;
}

__global__ void label_copy(const int* __restrict__ label, float* __restrict__ out)
{
    int i = blockIdx.x * 256 + threadIdx.x;
    if (i < BT * SQ) out[i] = (float)label[i];
}

__global__ void finalize(const float* __restrict__ accum, float* __restrict__ out)
{
    if (threadIdx.x == 0) {
        out[0] = -accum[0] / (float)BT;            // loss = -llh / B
        out[1] =  accum[1];                        // correct
    }
}

// ---------------------------------------------------------------------------
extern "C" void kernel_launch(void* const* d_in, const int* in_sizes, int n_in,
                              void* d_out, int out_size, void* d_ws, size_t ws_size,
                              hipStream_t stream)
{
    const float* hidden = (const float*)d_in[0];   // [64, 512, 768]
    const int*   label  = (const int*)  d_in[1];   // [64, 512]
    const int*   mask   = (const int*)  d_in[2];   // [64, 512]
    const float* W      = (const float*)d_in[3];   // [768, 9]
    const float* bias   = (const float*)d_in[4];   // [9]
    const float* startT = (const float*)d_in[5];   // [9]
    const float* endT   = (const float*)d_in[6];   // [9]
    const float* trans  = (const float*)d_in[7];   // [9, 9]

    float* out = (float*)d_out;                    // [loss, correct, predict(BS), label(BS)]

    float* accum = (float*)d_ws;                               // 2 floats
    float* emis  = (float*)((char*)d_ws + 256);                // [B*S, 9]

    init_accum<<<1, 32, 0, stream>>>(accum);

    // 32768 rows / (8 waves * 16 rows) = 256 blocks
    emis_gemm<<<256, 256, 0, stream>>>(hidden, W, bias, emis);

    // blocks [0,64): forward algorithm; blocks [64,128): Viterbi
    crf_scan<<<2 * BT, 32, 0, stream>>>(emis, label, mask, startT, endT, trans,
                                        out + 2, accum);

    label_copy<<<(BT * SQ + 255) / 256, 256, 0, stream>>>(label, out + 2 + BT * SQ);

    finalize<<<1, 32, 0, stream>>>(accum, out);
}
